// UnimpNet_30726196035831
// MI455X (gfx1250) — matrix operand order repaired and checked by reading, not compiled, not run on hardware
//
#include <hip/hip_runtime.h>
#include <hip/hip_bf16.h>
#include <math.h>

// ---------------------------------------------------------------------------
// UniMP forward for MI455X (gfx1250, wave32, WMMA)
//   N=100000 nodes, F=256, E=3.2M edges, HEADS=2, C=8 (HEADS*C = 16), 40 classes
// ---------------------------------------------------------------------------

typedef __bf16 bf16;
typedef __attribute__((ext_vector_type(16))) __bf16 v16bf;
typedef __attribute__((ext_vector_type(8)))  float  v8f;

#define FDIM   256
#define OUTC   16      // HEADS * C
#define NHEADS 2
#define LDS_STRIDE 264 // 256 + 8 bf16 pad -> breaks LDS bank conflicts on A gather
#define WAVES_PER_BLOCK 4

// order-preserving float <-> uint encoding for exact atomic max
__device__ __forceinline__ unsigned enc_f32(float f) {
  unsigned u = __float_as_uint(f);
  return (u & 0x80000000u) ? ~u : (u | 0x80000000u);
}
__device__ __forceinline__ float dec_f32(unsigned e) {
  unsigned u = (e & 0x80000000u) ? (e & 0x7FFFFFFFu) : ~e;
  return __uint_as_float(u);
}

// ---------------------------------------------------------------------------
// Pack the four F x 16 f32 weight matrices into bf16 WMMA B-fragment panels.
// Layout: wpack[g][kb][lane][j], g in {q,k,v,skip}, kb = K-block of 32.
// B 32x16 bf16 layout used here (contiguous k): lane<16 holds col N=lane, K = kb*32 + j;
//                          lane>=16 holds col N=lane-16, K = kb*32 + 16 + j.
// ---------------------------------------------------------------------------
__global__ void pack_weights_kernel(const float* __restrict__ Wq,
                                    const float* __restrict__ Wk,
                                    const float* __restrict__ Wv,
                                    const float* __restrict__ Ws,
                                    bf16* __restrict__ wpack) {
  int idx = blockIdx.x * blockDim.x + threadIdx.x;   // 4*8*32*16 = 16384
  if (idx >= 4 * 8 * 32 * 16) return;
  int j    = idx & 15;
  int lane = (idx >> 4) & 31;
  int kb   = (idx >> 9) & 7;
  int g    = idx >> 12;
  const float* W = (g == 0) ? Wq : (g == 1) ? Wk : (g == 2) ? Wv : Ws;
  int ncol = lane & 15;
  int krow = kb * 32 + ((lane >= 16) ? 16 : 0) + j;
  wpack[idx] = (bf16)W[krow * OUTC + ncol];
}

__global__ void init_stats_kernel(unsigned* __restrict__ amax,
                                  float* __restrict__ denom, int cnt) {
  int t = blockIdx.x * blockDim.x + threadIdx.x;
  if (t < cnt) { amax[t] = 0x007FFFFFu; /* enc(-inf) */ denom[t] = 0.0f; }
}

// ---------------------------------------------------------------------------
// Fused: h = x + mask*emb[y] (bf16 into LDS), then 4 projections via WMMA.
// One wave = one 16-row tile; 4 waves (128 threads) per block.
// skip projection (+bias) is written straight into the aggregation buffer.
// ---------------------------------------------------------------------------
__global__ __launch_bounds__(128) void fused_proj_kernel(
    const float* __restrict__ x, const int* __restrict__ y,
    const unsigned char* __restrict__ msk, const float* __restrict__ emb,
    const bf16* __restrict__ wpack,
    const float* __restrict__ bq, const float* __restrict__ bk,
    const float* __restrict__ bv, const float* __restrict__ bskip,
    float* __restrict__ q, float* __restrict__ k, float* __restrict__ v,
    float* __restrict__ agg, int n) {
  __shared__ bf16 hstage[WAVES_PER_BLOCK][16 * LDS_STRIDE];

  const int lane = threadIdx.x & 31;
  const int wave = threadIdx.x >> 5;
  const int tile = blockIdx.x * WAVES_PER_BLOCK + wave;
  const int row0 = tile * 16;
  const bool active = (row0 < n);
  bf16* hs = hstage[wave];

  if (active) {
    for (int r = 0; r < 16; ++r) {
      int node = row0 + r;
      if (node < n) {
        int cls = y[node];
        float m = msk[node] ? 1.0f : 0.0f;
        const float4* xr = (const float4*)(x + (size_t)node * FDIM);
        const float4* er = (const float4*)(emb + (size_t)cls * FDIM);
        #pragma unroll
        for (int i = lane; i < FDIM / 4; i += 32) {
          float4 xv = xr[i];
          float4 ev = er[i];
          int c = i * 4;
          hs[r * LDS_STRIDE + c + 0] = (bf16)(xv.x + m * ev.x);
          hs[r * LDS_STRIDE + c + 1] = (bf16)(xv.y + m * ev.y);
          hs[r * LDS_STRIDE + c + 2] = (bf16)(xv.z + m * ev.z);
          hs[r * LDS_STRIDE + c + 3] = (bf16)(xv.w + m * ev.w);
        }
      } else {
        for (int i = lane; i < FDIM; i += 32)
          hs[r * LDS_STRIDE + i] = (bf16)0.0f;
      }
    }
  }
  __syncthreads();   // uniform: makes per-wave LDS writes visible cross-lane
  if (!active) return;

  v8f accq = {0.f,0.f,0.f,0.f,0.f,0.f,0.f,0.f};
  v8f acck = {0.f,0.f,0.f,0.f,0.f,0.f,0.f,0.f};
  v8f accv = {0.f,0.f,0.f,0.f,0.f,0.f,0.f,0.f};
  v8f accs = {0.f,0.f,0.f,0.f,0.f,0.f,0.f,0.f};

  const int m16 = lane & 15;
  const int hi  = (lane >= 16) ? 8 : 0;

  #pragma unroll
  for (int kb = 0; kb < 8; ++kb) {
    // A fragment: 16-bit A 16x32, contiguous k (same order as B packing)
    //   lane L: row M = L&15, element j holds K = kb*32 + (L>=16?16:0) + j
    v16bf a;
    #pragma unroll
    for (int j = 0; j < 16; ++j) {
      int kk = kb * 32 + 2 * hi + j;
      a[j] = hs[m16 * LDS_STRIDE + kk];
    }
    // B fragments: contiguous 32B per lane from packed panels (L2-resident)
    const v16bf* w0 = (const v16bf*)(wpack + (size_t)(0 * 8 + kb) * 512);
    const v16bf* w1 = (const v16bf*)(wpack + (size_t)(1 * 8 + kb) * 512);
    const v16bf* w2 = (const v16bf*)(wpack + (size_t)(2 * 8 + kb) * 512);
    const v16bf* w3 = (const v16bf*)(wpack + (size_t)(3 * 8 + kb) * 512);
    v16bf b0 = w0[lane], b1 = w1[lane], b2 = w2[lane], b3 = w3[lane];

    accq = __builtin_amdgcn_wmma_f32_16x16x32_bf16(false, a, false, b0, (short)0, accq, false, false);
    acck = __builtin_amdgcn_wmma_f32_16x16x32_bf16(false, a, false, b1, (short)0, acck, false, false);
    accv = __builtin_amdgcn_wmma_f32_16x16x32_bf16(false, a, false, b2, (short)0, accv, false, false);
    accs = __builtin_amdgcn_wmma_f32_16x16x32_bf16(false, a, false, b3, (short)0, accs, false, false);
  }

  // C/D layout: VGPR r -> (M = r + hi, N = lane&15)
  const float biasq = bq[m16], biask = bk[m16], biasv = bv[m16], biass = bskip[m16];
  #pragma unroll
  for (int r = 0; r < 8; ++r) {
    int node = row0 + hi + r;
    if (node < n) {
      size_t o = (size_t)node * OUTC + m16;
      q[o]   = accq[r] + biasq;
      k[o]   = acck[r] + biask;
      v[o]   = accv[r] + biasv;
      agg[o] = accs[r] + biass;   // skip connection seeds the aggregation
    }
  }
}

// ---------------------------------------------------------------------------
// Edge pass A: per-edge attention logits (2 heads) + atomic segment max.
// ---------------------------------------------------------------------------
__global__ void edge_logits_kernel(const int* __restrict__ ei, int e,
                                   const float* __restrict__ q,
                                   const float* __restrict__ k,
                                   float* __restrict__ alpha,
                                   unsigned* __restrict__ amax) {
  int t = blockIdx.x * blockDim.x + threadIdx.x;
  if (t >= e) return;
  int src = ei[t];
  int dst = ei[e + t];
  const float4* qd = (const float4*)(q + (size_t)dst * OUTC);
  const float4* ks = (const float4*)(k + (size_t)src * OUTC);
  float4 q0 = qd[0], q1 = qd[1], q2 = qd[2], q3 = qd[3];
  float4 k0 = ks[0], k1 = ks[1], k2 = ks[2], k3 = ks[3];
  float a0 = q0.x*k0.x + q0.y*k0.y + q0.z*k0.z + q0.w*k0.w
           + q1.x*k1.x + q1.y*k1.y + q1.z*k1.z + q1.w*k1.w;
  float a1 = q2.x*k2.x + q2.y*k2.y + q2.z*k2.z + q2.w*k2.w
           + q3.x*k3.x + q3.y*k3.y + q3.z*k3.z + q3.w*k3.w;
  const float s = 0.35355339059327373f;  // 1/sqrt(C), C = 8
  a0 *= s; a1 *= s;
  ((float2*)alpha)[t] = make_float2(a0, a1);
  atomicMax(amax + 2 * (size_t)dst + 0, enc_f32(a0));
  atomicMax(amax + 2 * (size_t)dst + 1, enc_f32(a1));
}

// Edge pass B: exp(alpha - max) + atomic segment sum of denominators.
__global__ void edge_exp_kernel(const int* __restrict__ ei, int e,
                                const unsigned* __restrict__ amax,
                                float* __restrict__ alpha,
                                float* __restrict__ denom) {
  int t = blockIdx.x * blockDim.x + threadIdx.x;
  if (t >= e) return;
  int dst = ei[e + t];
  float2 a = ((float2*)alpha)[t];
  float e0 = __expf(a.x - dec_f32(amax[2 * (size_t)dst + 0]));
  float e1 = __expf(a.y - dec_f32(amax[2 * (size_t)dst + 1]));
  ((float2*)alpha)[t] = make_float2(e0, e1);
  atomicAdd(denom + 2 * (size_t)dst + 0, e0);
  atomicAdd(denom + 2 * (size_t)dst + 1, e1);
}

// Edge pass C: att = ex/denom; scatter-add att * v[src] into agg[dst].
__global__ void edge_agg_kernel(const int* __restrict__ ei, int e,
                                const float* __restrict__ alpha,
                                const float* __restrict__ denom,
                                const float* __restrict__ v,
                                float* __restrict__ agg) {
  int t = blockIdx.x * blockDim.x + threadIdx.x;
  if (t >= e) return;
  int src = ei[t];
  int dst = ei[e + t];
  float2 ex = ((const float2*)alpha)[t];
  float att0 = ex.x / (denom[2 * (size_t)dst + 0] + 1e-16f);
  float att1 = ex.y / (denom[2 * (size_t)dst + 1] + 1e-16f);
  const float4* vs = (const float4*)(v + (size_t)src * OUTC);
  float4 v0 = vs[0], v1 = vs[1], v2 = vs[2], v3 = vs[3];
  float* d = agg + (size_t)dst * OUTC;
  atomicAdd(d + 0,  v0.x * att0); atomicAdd(d + 1,  v0.y * att0);
  atomicAdd(d + 2,  v0.z * att0); atomicAdd(d + 3,  v0.w * att0);
  atomicAdd(d + 4,  v1.x * att0); atomicAdd(d + 5,  v1.y * att0);
  atomicAdd(d + 6,  v1.z * att0); atomicAdd(d + 7,  v1.w * att0);
  atomicAdd(d + 8,  v2.x * att1); atomicAdd(d + 9,  v2.y * att1);
  atomicAdd(d + 10, v2.z * att1); atomicAdd(d + 11, v2.w * att1);
  atomicAdd(d + 12, v3.x * att1); atomicAdd(d + 13, v3.y * att1);
  atomicAdd(d + 14, v3.z * att1); atomicAdd(d + 15, v3.w * att1);
}

// Final classifier: out[n, cls] = agg[n, :] . Wout[:, cls] + bout[cls]
__global__ void classify_kernel(const float* __restrict__ agg,
                                const float* __restrict__ Wout,
                                const float* __restrict__ bout,
                                float* __restrict__ out, int n, int nc) {
  int t = blockIdx.x * blockDim.x + threadIdx.x;
  if (t >= n * nc) return;
  int node = t / nc;
  int cls  = t - node * nc;
  const float* ar = agg + (size_t)node * OUTC;
  float s = bout[cls];
  #pragma unroll
  for (int j = 0; j < OUTC; ++j) s = fmaf(ar[j], Wout[j * nc + cls], s);
  out[t] = s;
}

// ---------------------------------------------------------------------------
extern "C" void kernel_launch(void* const* d_in, const int* in_sizes, int n_in,
                              void* d_out, int out_size, void* d_ws, size_t ws_size,
                              hipStream_t stream) {
  const float*         x     = (const float*)d_in[0];
  const int*           y     = (const int*)d_in[1];
  const int*           ei    = (const int*)d_in[2];
  const unsigned char* msk   = (const unsigned char*)d_in[3];
  const float*         emb   = (const float*)d_in[4];
  const float*         Wq    = (const float*)d_in[5];
  const float*         bq    = (const float*)d_in[6];
  const float*         Wk    = (const float*)d_in[7];
  const float*         bk    = (const float*)d_in[8];
  const float*         Wv    = (const float*)d_in[9];
  const float*         bv    = (const float*)d_in[10];
  const float*         Wskip = (const float*)d_in[11];
  const float*         bskip = (const float*)d_in[12];
  const float*         Wout  = (const float*)d_in[13];
  const float*         bout  = (const float*)d_in[14];

  const int n  = in_sizes[0] / FDIM;   // 100000
  const int e  = in_sizes[2] / 2;      // 3200000
  const int nc = in_sizes[14];         // 40

  char* ws = (char*)d_ws;
  size_t off = 0;
  auto alloc = [&](size_t bytes) -> char* {
    char* p = ws + off;
    off += (bytes + 255) & ~(size_t)255;
    return p;
  };
  float*    q     = (float*)alloc((size_t)n * OUTC * sizeof(float));
  float*    k     = (float*)alloc((size_t)n * OUTC * sizeof(float));
  float*    v     = (float*)alloc((size_t)n * OUTC * sizeof(float));
  float*    agg   = (float*)alloc((size_t)n * OUTC * sizeof(float));
  unsigned* amax  = (unsigned*)alloc((size_t)n * NHEADS * sizeof(unsigned));
  float*    denom = (float*)alloc((size_t)n * NHEADS * sizeof(float));
  float*    alpha = (float*)alloc((size_t)e * NHEADS * sizeof(float));
  bf16*     wpack = (bf16*)alloc((size_t)4 * 8 * 512 * sizeof(bf16));

  pack_weights_kernel<<<(4 * 8 * 512 + 255) / 256, 256, 0, stream>>>(
      Wq, Wk, Wv, Wskip, wpack);
  init_stats_kernel<<<(n * NHEADS + 255) / 256, 256, 0, stream>>>(
      amax, denom, n * NHEADS);

  const int tiles = (n + 15) / 16;
  fused_proj_kernel<<<(tiles + WAVES_PER_BLOCK - 1) / WAVES_PER_BLOCK, 128, 0, stream>>>(
      x, y, msk, emb, wpack, bq, bk, bv, bskip, q, k, v, agg, n);

  const int eblocks = (e + 255) / 256;
  edge_logits_kernel<<<eblocks, 256, 0, stream>>>(ei, e, q, k, alpha, amax);
  edge_exp_kernel<<<eblocks, 256, 0, stream>>>(ei, e, amax, alpha, denom);
  edge_agg_kernel<<<eblocks, 256, 0, stream>>>(ei, e, alpha, denom, v, agg);

  classify_kernel<<<((size_t)n * nc + 255) / 256, 256, 0, stream>>>(
      agg, Wout, bout, (float*)d_out, n, nc);
}
